// BEVFormerEncoderLayer_21741124453156
// MI455X (gfx1250) — matrix operand order repaired and checked by reading, not compiled, not run on hardware
//
#include <hip/hip_runtime.h>
#include <hip/hip_bf16.h>
#include <stdint.h>

// ---------------------------------------------------------------------------
// BEVFormer encoder layer for MI455X (gfx1250), wave32 + WMMA bf16.
// Round 4: 1/sqrt(d) folded into Q projection weights/bias; flash-attn
// denominator kept as per-lane partials (one epilogue reduction instead of
// per-chunk); mask-free main chunk loop + single masked tail chunk.
// ---------------------------------------------------------------------------

typedef __attribute__((ext_vector_type(16))) __bf16       v16bf;
typedef __attribute__((ext_vector_type(8)))  float        v8f;
typedef __attribute__((ext_vector_type(4)))  unsigned int v4u;

#define NQ   2500      // BEV queries (50x50)
#define NKV  5000      // kv rows = query + aligned prev
#define CDIM 256
#define HEADS 8
#define HD   32        // head dim
#define HFI  28        // img feat H
#define WFI  50        // img feat W
#define LDVT 5024      // padded key stride for transposed V (multiple of 32)

__device__ __forceinline__ unsigned short f2b(float f) {
  union { float f; unsigned u; } x; x.f = f;
  unsigned r = x.u + 0x7FFFu + ((x.u >> 16) & 1u);  // RNE fp32 -> bf16
  return (unsigned short)(r >> 16);
}

union FragU { v4u d[2]; v16bf b; };

// A-matrix 16x32 bf16 fragment (ISA 7.12.2). `rowPtr` is the per-lane,
// pre-clamped pointer  A + row*lda + half*16  (row clamped into [0,M)).
__device__ __forceinline__ v16bf load_a_frag_ptr(const unsigned short* rowPtr, int kb) {
  FragU f;
  f.d[0] = *(const v4u*)(rowPtr + kb);        // K offsets +0..7  (this half)
  f.d[1] = *(const v4u*)(rowPtr + kb + 8);    // K offsets +8..15
  return f.b;
}

// B-matrix 32x16 bf16 fragment from row-major W[N][K] (computes A @ W^T).
// `colPtr` is the per-lane, pre-clamped pointer  W + col*ldw + half*16.
__device__ __forceinline__ v16bf load_b_frag_ptr(const unsigned short* colPtr, int kb) {
  FragU f;
  f.d[0] = *(const v4u*)(colPtr + kb);
  f.d[1] = *(const v4u*)(colPtr + kb + 8);
  return f.b;
}

__device__ __forceinline__ v8f wmma_bf16(v16bf a, v16bf b, v8f c) {
  return __builtin_amdgcn_wmma_f32_16x16x32_bf16(false, a, false, b,
                                                 (short)0, c, false, false);
}

// ---- DPP16 16-lane butterfly reductions (VALU only, no LDS) ----------------
template <int CTRL>
__device__ __forceinline__ float dpp_movf(float v) {
  return __builtin_bit_cast(float,
      __builtin_amdgcn_update_dpp(0, __builtin_bit_cast(int, v),
                                  CTRL, 0xF, 0xF, true));
}
__device__ __forceinline__ float red16_max(float x) {
  x = fmaxf(x, dpp_movf<0xB1>(x));    // quad_perm(1,0,3,2)
  x = fmaxf(x, dpp_movf<0x4E>(x));    // quad_perm(2,3,0,1)
  x = fmaxf(x, dpp_movf<0x141>(x));   // row_half_mirror
  x = fmaxf(x, dpp_movf<0x140>(x));   // row_mirror
  return x;
}
__device__ __forceinline__ float red16_sum(float x) {
  x += dpp_movf<0xB1>(x);
  x += dpp_movf<0x4E>(x);
  x += dpp_movf<0x141>(x);
  x += dpp_movf<0x140>(x);
  return x;
}

// ---------------------------------------------------------------------------
// Generic bf16 WMMA GEMM: out = act( A[MxK] @ W[NxK]^T + bias + residual )
// One 16x16 output tile per wave, 8 waves per block. K fully unrolled.
// OOB rows/cols are clamped (never stored), so the K-loop is branch-free.
// ---------------------------------------------------------------------------
template <int KT>
__global__ void gemm_bf16(const unsigned short* __restrict__ A, int lda,
                          const unsigned short* __restrict__ W, int ldw,
                          const float* __restrict__ bias,
                          const float* __restrict__ residual,
                          float* __restrict__ outF,
                          unsigned short* __restrict__ outB,
                          unsigned short* __restrict__ outBT, int ldbt,
                          int M, int N, int ldc, int relu) {
  int wid    = blockIdx.x * (blockDim.x >> 5) + (threadIdx.x >> 5);
  int tilesN = (N + 15) >> 4;
  int tilesM = (M + 15) >> 4;
  if (wid >= tilesM * tilesN) return;
  int row0 = (wid / tilesN) * 16;
  int col0 = (wid % tilesN) * 16;

  int lane = threadIdx.x & 31;
  int half = lane >> 4;
  int l15  = lane & 15;
  int arow = min(row0 + l15, M - 1);            // clamp: OOB rows never stored
  int bcol = min(col0 + l15, N - 1);            // clamp: OOB cols never stored
  const unsigned short* aPtr = A + (size_t)arow * lda + half * 16;
  const unsigned short* bPtr = W + (size_t)bcol * ldw + half * 16;

  v8f acc;
#pragma unroll
  for (int i = 0; i < 8; ++i) acc[i] = 0.f;

#pragma unroll
  for (int kb = 0; kb < KT; kb += 32) {
    v16bf a = load_a_frag_ptr(aPtr, kb);
    v16bf b = load_b_frag_ptr(bPtr, kb);
    acc = wmma_bf16(a, b, acc);
  }

  int col = col0 + l15;
#pragma unroll
  for (int r = 0; r < 8; ++r) {
    int row = row0 + half * 8 + r;   // C layout: VGPR r -> row r (+8 for hi half)
    if (row < M && col < N) {
      float v = acc[r];
      if (bias)     v += bias[col];
      if (residual) v += residual[(size_t)row * ldc + col];
      if (relu)     v = fmaxf(v, 0.f);
      if (outF)  outF[(size_t)row * ldc + col] = v;
      if (outB)  outB[(size_t)row * ldc + col] = f2b(v);
      if (outBT) outBT[(size_t)col * ldbt + row] = f2b(v);  // transposed (for V)
    }
  }
}

// ---------------------------------------------------------------------------
// One 32-key flash-attention chunk. Q is pre-scaled by 1/sqrt(d), so raw
// WMMA scores are final logits. MASK=true only for the final partial chunk.
// ---------------------------------------------------------------------------
template <bool MASK>
__device__ __forceinline__ void attn_chunk(
    int kk, int h, int nn, int half,
    const unsigned short* __restrict__ Kb,
    unsigned short* Pt, const unsigned short* pPtr,
    const unsigned short* v1Ptr, const unsigned short* v2Ptr,
    v16bf qf, float (&m_r)[8], float (&l_r)[8], v8f& o1, v8f& o2) {
  int key1 = MASK ? min(kk + nn, NKV - 1)      : (kk + nn);
  int key2 = MASK ? min(kk + 16 + nn, NKV - 1) : (kk + 16 + nn);
  const unsigned short* k1Ptr = Kb + (size_t)key1 * CDIM + h * HD + half * 16;
  const unsigned short* k2Ptr = Kb + (size_t)key2 * CDIM + h * HD + half * 16;
  v16bf k1 = load_b_frag_ptr(k1Ptr, 0);
  v16bf k2 = load_b_frag_ptr(k2Ptr, 0);
  v8f z;
#pragma unroll
  for (int i = 0; i < 8; ++i) z[i] = 0.f;
  v8f s1 = wmma_bf16(qf, k1, z);     // logits, keys kk..kk+15
  v8f s2 = wmma_bf16(qf, k2, z);     // logits, keys kk+16..kk+31
  bool ok1 = !MASK || (kk + nn) < NKV;
  bool ok2 = !MASK || (kk + 16 + nn) < NKV;

#pragma unroll
  for (int r = 0; r < 8; ++r) {
    float a = MASK ? (ok1 ? s1[r] : -1e30f) : s1[r];
    float b = MASK ? (ok2 ? s2[r] : -1e30f) : s2[r];
    float mx = red16_max(fmaxf(a, b));   // row max across this half's 16 lanes
    float nm = fmaxf(m_r[r], mx);
    float e1 = __expf(a - nm);
    float e2 = __expf(b - nm);
    float alpha = __expf(m_r[r] - nm);   // row-uniform within the half
    l_r[r] = l_r[r] * alpha + (e1 + e2); // per-lane partial row sum (deferred)
    m_r[r] = nm;
    o1[r] *= alpha;
    o2[r] *= alpha;
    int prow = half * 8 + r;
    Pt[prow * 32 + nn]      = f2b(e1);
    Pt[prow * 32 + 16 + nn] = f2b(e2);
  }
  __syncthreads();
  v16bf pf = load_a_frag_ptr(pPtr, 0);               // P as A-matrix
  v16bf v1 = load_b_frag_ptr(v1Ptr, kk);
  v16bf v2 = load_b_frag_ptr(v2Ptr, kk);
  o1 = wmma_bf16(pf, v1, o1);
  o2 = wmma_bf16(pf, v2, o2);
  __syncthreads();
}

// ---------------------------------------------------------------------------
// Flash attention: one wave per (16-query tile, head).
// ---------------------------------------------------------------------------
__global__ void flash_attn(const unsigned short* __restrict__ Qb,
                           const unsigned short* __restrict__ Kb,
                           const unsigned short* __restrict__ Vt,
                           unsigned short* __restrict__ attnB) {
  int tq   = blockIdx.x;
  int h    = blockIdx.y;
  int row0 = tq * 16;
  int lane = threadIdx.x & 31;
  int half = lane >> 4;
  int nn   = lane & 15;
  __shared__ __align__(16) unsigned short Pt[16 * 32];

  // Q fragment (row-clamped; OOB query rows produce garbage scores that are
  // never stored and cannot contaminate valid rows).
  int qrow = min(row0 + nn, NQ - 1);
  const unsigned short* qPtr = Qb + (size_t)qrow * CDIM + h * HD + half * 16;
  v16bf qf = load_a_frag_ptr(qPtr, 0);

  // V fragments: per-lane base pointers (cols always < 256; Vt key-padded).
  const unsigned short* v1Ptr = Vt + (size_t)(h * HD + nn)      * LDVT + half * 16;
  const unsigned short* v2Ptr = Vt + (size_t)(h * HD + 16 + nn) * LDVT + half * 16;

  // P fragment source in LDS (always fully valid).
  const unsigned short* pPtr = Pt + (size_t)nn * 32 + half * 16;

  float m_r[8], l_r[8];
  v8f o1, o2;
#pragma unroll
  for (int r = 0; r < 8; ++r) { m_r[r] = -1e30f; l_r[r] = 0.f; o1[r] = 0.f; o2[r] = 0.f; }

  constexpr int KFULL = NKV & ~31;   // 4992: chunks needing no key masking
  for (int kk = 0; kk < KFULL; kk += 32)
    attn_chunk<false>(kk, h, nn, half, Kb, Pt, pPtr, v1Ptr, v2Ptr,
                      qf, m_r, l_r, o1, o2);
  attn_chunk<true>(KFULL, h, nn, half, Kb, Pt, pPtr, v1Ptr, v2Ptr,
                   qf, m_r, l_r, o1, o2);

#pragma unroll
  for (int r = 0; r < 8; ++r) {
    int row = row0 + half * 8 + r;
    if (row < NQ) {
      float inv = 1.0f / red16_sum(l_r[r]);   // single deferred denominator
      attnB[(size_t)row * CDIM + h * HD + nn]      = f2b(o1[r] * inv);
      attnB[(size_t)row * CDIM + h * HD + 16 + nn] = f2b(o2[r] * inv);
    }
  }
}

// ---------------------------------------------------------------------------
// Warp prev_bev by ego motion (affine grid sample) and pack kv to bf16.
// kv rows [0,NQ) = query, rows [NQ,2NQ) = aligned prev.
// ---------------------------------------------------------------------------
__global__ void prep_warp(const float* __restrict__ query,
                          const float* __restrict__ prev,
                          const float* __restrict__ ego,
                          unsigned short* __restrict__ kvB) {
  int q = blockIdx.x;
  int c = threadIdx.x;
  kvB[(size_t)q * CDIM + c] = f2b(query[(size_t)q * CDIM + c]);

  float tx = ego[3], ty = ego[7];
  float yaw = atan2f(ego[4], ego[0]);
  float dx = tx * 0.01953125f;   // tx / (102.4/50) * (2/50)
  float dy = ty * 0.01953125f;
  float co = cosf(yaw), si = sinf(yaw);
  int iy = q / 50, ix = q % 50;
  float bx = (2.f * ix + 1.f) / 50.f - 1.f;
  float by = (2.f * iy + 1.f) / 50.f - 1.f;
  float u = co * bx - si * by + dx;
  float v = si * bx + co * by + dy;
  float x = ((u + 1.f) * 50.f - 1.f) * 0.5f;
  float y = ((v + 1.f) * 50.f - 1.f) * 0.5f;
  float fx = floorf(x), fy = floorf(y);
  float wx = x - fx, wy = y - fy;
  int x0 = (int)fx, y0 = (int)fy;
  auto g = [&](int xi, int yi) -> float {
    if (xi < 0 || xi >= 50 || yi < 0 || yi >= 50) return 0.f;
    return prev[((size_t)(yi * 50 + xi)) * CDIM + c];
  };
  float acc = g(x0, y0) * (1.f - wx) * (1.f - wy) + g(x0 + 1, y0) * wx * (1.f - wy)
            + g(x0, y0 + 1) * (1.f - wx) * wy + g(x0 + 1, y0 + 1) * wx * wy;
  kvB[(size_t)(NQ + q) * CDIM + c] = f2b(acc);
}

__global__ void cvt_bf16_k(const float* __restrict__ src,
                           unsigned short* __restrict__ dst, int n, float scale) {
  int i = blockIdx.x * blockDim.x + threadIdx.x;
  if (i < n) dst[i] = f2b(src[i] * scale);
}

__global__ void scale_f32_k(const float* __restrict__ src,
                            float* __restrict__ dst, int n, float scale) {
  int i = blockIdx.x * blockDim.x + threadIdx.x;
  if (i < n) dst[i] = src[i] * scale;
}

__global__ void fill_u16(unsigned short* __restrict__ p, int n, unsigned short v) {
  int i = blockIdx.x * blockDim.x + threadIdx.x;
  if (i < n) p[i] = v;
}

// ---------------------------------------------------------------------------
// out = LayerNorm(x [+ y]) * g + b [+ post]. One block (256 thr) per row.
// ---------------------------------------------------------------------------
__global__ void add_ln(const float* __restrict__ x, const float* __restrict__ y,
                       const float* __restrict__ g, const float* __restrict__ b,
                       const float* __restrict__ post,
                       float* __restrict__ outF, unsigned short* __restrict__ outB) {
  int row = blockIdx.x, c = threadIdx.x;
  __shared__ float red[CDIM];
  size_t idx = (size_t)row * CDIM + c;
  float v = x[idx] + (y ? y[idx] : 0.f);
  red[c] = v; __syncthreads();
  for (int s = 128; s > 0; s >>= 1) { if (c < s) red[c] += red[c + s]; __syncthreads(); }
  float mean = red[0] * (1.f / CDIM);
  __syncthreads();
  float d = v - mean;
  red[c] = d * d; __syncthreads();
  for (int s = 128; s > 0; s >>= 1) { if (c < s) red[c] += red[c + s]; __syncthreads(); }
  float var = red[0] * (1.f / CDIM);
  float r = d * rsqrtf(var + 1e-5f) * g[c] + b[c];
  if (post) r += post[idx];
  if (outF) outF[idx] = r;
  if (outB) outB[idx] = f2b(r);
}

// ---------------------------------------------------------------------------
// Spatial cross-attention sampling: block per query, thread per channel.
// agg[q,c] = sum_{n,z} w[q,n,z] * bilinear(img[n,c], proj(q,n,z))
// ---------------------------------------------------------------------------
__global__ void sca_sample(const float* __restrict__ wlog,   // [NQ,24] logits
                           const float* __restrict__ img,    // [6,256,28,50]
                           const float* __restrict__ l2i,    // [6,4,4]
                           unsigned short* __restrict__ aggB) {
  int q = blockIdx.x;
  int c = threadIdx.x;
  __shared__ float sw[24], swx[24], swy[24];
  __shared__ int   sx0[24], sy0[24], svalid[24];
  if (c < 24) {
    float mx = -1e30f;
    for (int j = 0; j < 24; ++j) mx = fmaxf(mx, wlog[q * 24 + j]);
    float s = 0.f;
    for (int j = 0; j < 24; ++j) s += __expf(wlog[q * 24 + j] - mx);
    sw[c] = __expf(wlog[q * 24 + c] - mx) / s;

    int n = c >> 2, z = c & 3;
    int iy = q / 50, ix = q % 50;
    float X = ((ix + 0.5f) / 50.f) * 102.4f - 51.2f;
    float Y = ((iy + 0.5f) / 50.f) * 102.4f - 51.2f;
    float Z = ((z + 0.5f) / 4.f) * 8.f - 5.f;
    const float* Mt = l2i + n * 16;
    float cx = Mt[0] * X + Mt[1] * Y + Mt[2]  * Z + Mt[3];
    float cy = Mt[4] * X + Mt[5] * Y + Mt[6]  * Z + Mt[7];
    float cd = Mt[8] * X + Mt[9] * Y + Mt[10] * Z + Mt[11];
    float dep = fmaxf(cd, 1e-5f);
    float gx = (cx / dep / (WFI - 1.f) - 0.5f) * 2.f;
    float gy = (cy / dep / (HFI - 1.f) - 0.5f) * 2.f;
    svalid[c] = (cd > 1e-5f) && (gx > -1.f) && (gx < 1.f) && (gy > -1.f) && (gy < 1.f);
    float sxf = ((gx + 1.f) * WFI - 1.f) * 0.5f;
    float syf = ((gy + 1.f) * HFI - 1.f) * 0.5f;
    float fx = floorf(sxf), fy = floorf(syf);
    sx0[c] = (int)fx; sy0[c] = (int)fy;
    swx[c] = sxf - fx; swy[c] = syf - fy;
  }
  __syncthreads();

  float acc = 0.f;
  for (int s = 0; s < 24; ++s) {
    if (!svalid[s]) continue;   // uniform across block
    int n = s >> 2;
    const float* plane = img + ((size_t)n * CDIM + c) * (HFI * WFI);
    int x0 = sx0[s], y0 = sy0[s];
    float wx = swx[s], wy = swy[s];
    auto g = [&](int xi, int yi) -> float {
      if (xi < 0 || xi >= WFI || yi < 0 || yi >= HFI) return 0.f;
      return plane[yi * WFI + xi];
    };
    float sv = g(x0, y0) * (1.f - wx) * (1.f - wy) + g(x0 + 1, y0) * wx * (1.f - wy)
             + g(x0, y0 + 1) * (1.f - wx) * wy + g(x0 + 1, y0 + 1) * wx * wy;
    acc += sv * sw[s];
  }
  aggB[(size_t)q * CDIM + c] = f2b(acc);
}

// ---------------------------------------------------------------------------
extern "C" void kernel_launch(void* const* d_in, const int* in_sizes, int n_in,
                              void* d_out, int out_size, void* d_ws, size_t ws_size,
                              hipStream_t stream) {
  (void)in_sizes; (void)n_in; (void)out_size; (void)ws_size;
  const float* query     = (const float*)d_in[0];
  const float* prev      = (const float*)d_in[1];
  const float* imgf      = (const float*)d_in[2];
  const float* ego       = (const float*)d_in[3];
  const float* l2i       = (const float*)d_in[4];
  const float* attn_w_W  = (const float*)d_in[5];
  const float* attn_w_b  = (const float*)d_in[6];
  const float* sca_out_W = (const float*)d_in[7];
  const float* sca_out_b = (const float*)d_in[8];
  const float* in_proj_W = (const float*)d_in[9];
  const float* in_proj_b = (const float*)d_in[10];
  const float* mha_out_W = (const float*)d_in[11];
  const float* mha_out_b = (const float*)d_in[12];
  const float* n1g       = (const float*)d_in[13];
  const float* n1b       = (const float*)d_in[14];
  const float* ffn_W1    = (const float*)d_in[15];
  const float* ffn_b1    = (const float*)d_in[16];
  const float* ffn_W2    = (const float*)d_in[17];
  const float* ffn_b2    = (const float*)d_in[18];
  const float* n2g       = (const float*)d_in[19];
  const float* n2b       = (const float*)d_in[20];

  const float qscale = 0.17677669529663689f;  // 1/sqrt(32)

  char* base = (char*)d_ws;
  size_t off = 0;
  auto take = [&](size_t bytes) -> void* {
    off = (off + 255) & ~(size_t)255;
    void* p = base + off;
    off += bytes;
    return p;
  };
  unsigned short* kvB   = (unsigned short*)take((size_t)NKV * CDIM * 2);
  unsigned short* wQKV  = (unsigned short*)take((size_t)3 * CDIM * CDIM * 2);
  unsigned short* wOut  = (unsigned short*)take((size_t)CDIM * CDIM * 2);
  unsigned short* wSca  = (unsigned short*)take((size_t)CDIM * CDIM * 2);
  unsigned short* wAw   = (unsigned short*)take((size_t)24 * CDIM * 2);
  unsigned short* wF1   = (unsigned short*)take((size_t)2 * CDIM * CDIM * 2);
  unsigned short* wF2   = (unsigned short*)take((size_t)2 * CDIM * CDIM * 2);
  float*          qbF   = (float*)take((size_t)CDIM * 4);   // scaled Q bias
  unsigned short* Qb    = (unsigned short*)take((size_t)NQ * CDIM * 2);
  unsigned short* Kb    = (unsigned short*)take((size_t)NKV * CDIM * 2);
  unsigned short* Vt    = (unsigned short*)take((size_t)CDIM * LDVT * 2);
  unsigned short* attnB = (unsigned short*)take((size_t)NQ * CDIM * 2);
  float*          projF = (float*)take((size_t)NQ * CDIM * 4);
  float*          qF    = (float*)take((size_t)NQ * CDIM * 4);
  unsigned short* qB    = (unsigned short*)take((size_t)NQ * CDIM * 2);
  float*          wlogF = (float*)take((size_t)NQ * 24 * 4);
  unsigned short* aggB  = (unsigned short*)take((size_t)NQ * CDIM * 2);
  float*          q2F   = (float*)take((size_t)NQ * CDIM * 4);
  unsigned short* q2B   = (unsigned short*)take((size_t)NQ * CDIM * 2);
  unsigned short* h1B   = (unsigned short*)take((size_t)NQ * 2 * CDIM * 2);
  float*          hF    = (float*)take((size_t)NQ * CDIM * 4);

  // 1) warp prev_bev + pack kv to bf16
  prep_warp<<<NQ, CDIM, 0, stream>>>(query, prev, ego, kvB);

  // 2) weight conversions fp32 -> bf16 (Q block pre-scaled by 1/sqrt(d))
  auto cvt = [&](const float* s, unsigned short* dptr, int n, float sc) {
    cvt_bf16_k<<<(n + 255) / 256, 256, 0, stream>>>(s, dptr, n, sc);
  };
  cvt(in_proj_W,               wQKV,               CDIM * CDIM,     qscale);
  cvt(in_proj_W + CDIM * CDIM, wQKV + CDIM * CDIM, 2 * CDIM * CDIM, 1.f);
  cvt(mha_out_W, wOut, CDIM * CDIM,     1.f);
  cvt(sca_out_W, wSca, CDIM * CDIM,     1.f);
  cvt(attn_w_W,  wAw,  24 * CDIM,       1.f);
  cvt(ffn_W1,    wF1,  2 * CDIM * CDIM, 1.f);
  cvt(ffn_W2,    wF2,  2 * CDIM * CDIM, 1.f);
  scale_f32_k<<<1, CDIM, 0, stream>>>(in_proj_b, qbF, CDIM, qscale);
  fill_u16<<<(CDIM * LDVT + 255) / 256, 256, 0, stream>>>(Vt, CDIM * LDVT, (unsigned short)0);

  auto gemm256 = [&](const unsigned short* A, int lda, const unsigned short* W, int ldw,
                     const float* bias, const float* residual, float* oF,
                     unsigned short* oB, unsigned short* oBT, int ldbt,
                     int M, int N, int ldc, int relu) {
    int tiles = ((M + 15) / 16) * ((N + 15) / 16);
    gemm_bf16<256><<<(tiles + 7) / 8, 256, 0, stream>>>(A, lda, W, ldw, bias, residual,
                                                        oF, oB, oBT, ldbt, M, N, ldc, relu);
  };
  auto gemm512 = [&](const unsigned short* A, int lda, const unsigned short* W, int ldw,
                     const float* bias, const float* residual, float* oF,
                     unsigned short* oB, unsigned short* oBT, int ldbt,
                     int M, int N, int ldc, int relu) {
    int tiles = ((M + 15) / 16) * ((N + 15) / 16);
    gemm_bf16<512><<<(tiles + 7) / 8, 256, 0, stream>>>(A, lda, W, ldw, bias, residual,
                                                        oF, oB, oBT, ldbt, M, N, ldc, relu);
  };

  // 3) Q / K / V projections (Q pre-scaled; V written transposed)
  gemm256(kvB, CDIM, wQKV,                   CDIM, qbF,                  nullptr,
          nullptr, Qb, nullptr, 0, NQ,  CDIM, CDIM, 0);
  gemm256(kvB, CDIM, wQKV + CDIM * CDIM,     CDIM, in_proj_b + CDIM,     nullptr,
          nullptr, Kb, nullptr, 0, NKV, CDIM, CDIM, 0);
  gemm256(kvB, CDIM, wQKV + 2 * CDIM * CDIM, CDIM, in_proj_b + 2 * CDIM, nullptr,
          nullptr, nullptr, Vt, LDVT, NKV, CDIM, CDIM, 0);

  // 4) flash attention, one wave per (query tile, head)
  dim3 agrid((NQ + 15) / 16, HEADS);
  flash_attn<<<agrid, 32, 0, stream>>>(Qb, Kb, Vt, attnB);

  // 5) MHA out projection, then q = LN(query + attn)
  gemm256(attnB, CDIM, wOut, CDIM, mha_out_b, nullptr, projF, nullptr, nullptr, 0,
          NQ, CDIM, CDIM, 0);
  add_ln<<<NQ, CDIM, 0, stream>>>(projF, query, n1g, n1b, nullptr, qF, qB);

  // 6) camera/level attention-weight logits (N=24), then gather-aggregate
  gemm256(qB, CDIM, wAw, CDIM, attn_w_b, nullptr, wlogF, nullptr, nullptr, 0,
          NQ, 24, 24, 0);
  sca_sample<<<NQ, CDIM, 0, stream>>>(wlogF, imgf, l2i, aggB);

  // 7) q2 = agg @ sca_out^T + b + q
  gemm256(aggB, CDIM, wSca, CDIM, sca_out_b, qF, q2F, q2B, nullptr, 0,
          NQ, CDIM, CDIM, 0);

  // 8) FFN: h = relu(q2 @ W1^T + b1) @ W2^T + b2
  gemm256(q2B, CDIM,     wF1, CDIM,     ffn_b1, nullptr, nullptr, h1B, nullptr, 0,
          NQ, 2 * CDIM, 2 * CDIM, 1);
  gemm512(h1B, 2 * CDIM, wF2, 2 * CDIM, ffn_b2, nullptr, hF, nullptr, nullptr, 0,
          NQ, CDIM,     CDIM,     0);

  // 9) out = q2 + LN(h)
  add_ln<<<NQ, CDIM, 0, stream>>>(hF, nullptr, n2g, n2b, q2F, (float*)d_out, nullptr);
}
